// Transformation_16509854286115
// MI455X (gfx1250) — matrix operand order repaired and checked by reading, not compiled, not run on hardware
//
#include <hip/hip_runtime.h>
#include <math.h>

typedef __attribute__((ext_vector_type(16))) _Float16 v16h;
typedef __attribute__((ext_vector_type(8)))  _Float16 v8h;
typedef __attribute__((ext_vector_type(8)))  float    v8f;

#define DEG 10

// Basis -> WMMA for one group of 16 flat elements.
// B-matrix col j = element; lane j fills K=0..10 with x^v*(1-x)^(10-v),
// lane 16+j fills K=16..25 with x^v'*(1-x)^(9-v'). Binomials/params live in A.
__device__ __forceinline__ v8f bern_group(float x, bool hiHalf, v16h afrag) {
    const float y = 1.0f - x;
    float xp[DEG + 1], yp[DEG + 1];
    xp[0] = 1.0f; yp[0] = 1.0f;
#pragma unroll
    for (int v = 1; v <= DEG; ++v) { xp[v] = xp[v - 1] * x; yp[v] = yp[v - 1] * y; }

    float s[16];
#pragma unroll
    for (int v = 0; v <= 9; ++v) s[v] = xp[v] * (hiHalf ? yp[9 - v] : yp[10 - v]);
    s[10] = hiHalf ? 0.0f : xp[10];
    s[11] = s[12] = s[13] = s[14] = s[15] = 0.0f;

    v16h bfrag;
#pragma unroll
    for (int v = 0; v < 16; ++v) bfrag[v] = (_Float16)s[v];

    v8f c = {0.f, 0.f, 0.f, 0.f, 0.f, 0.f, 0.f, 0.f};
    return __builtin_amdgcn_wmma_f32_16x16x32_f16(
        /*neg_a=*/false, afrag, /*neg_b=*/false, bfrag,
        /*c_mod=*/(short)0, c, /*reuse_a=*/false, /*reuse_b=*/false);
}

// D rows: 0-2 outputs, 3-5 derivs (rows 8-13 duplicate so both halves see all 6).
__device__ __forceinline__ float bern_extract(v8f d, int k, bool hiHalf, float ldglob) {
    const bool k1 = (k == 1), k2 = (k == 2);
    const float outv = k2 ? d[2] : (k1 ? d[1] : d[0]);
    const float derv = k2 ? d[5] : (k1 ? d[4] : d[3]);
    const float lg   = ldglob + 0.6931471805599453f * __log2f(fabsf(derv));
    return hiHalf ? lg : outv;
}

__global__ __launch_bounds__(256)
void Transformation_16509854286115_kernel(const float* __restrict__ inp,
                                          const float* __restrict__ params,
                                          const float* __restrict__ prange,
                                          const float* __restrict__ ldg_ptr,
                                          float* __restrict__ out,
                                          int ntot, int ngroups, int gpair) {
    __shared__ __align__(16) _Float16 sA[16 * 32];

    // ---- build constant A matrix cooperatively (one f16 pair per thread) ----
    {
        const int t   = threadIdx.x;       // 0..255
        const int row = t >> 4;            // 0..15
        const int K0  = (t & 15) << 1;     // 0,2,...,30
        float v0 = 0.0f, v1 = 0.0f;
        const int ch = (row < 6) ? row : ((row >= 8 && row < 14) ? (row - 8) : -1);
        if (ch >= 0) {
            const int i = (ch < 3) ? ch : (ch - 3);
            float pr[DEG + 1];
            pr[0] = params[i];
#pragma unroll
            for (int v = 1; v <= DEG; ++v) pr[v] = pr[v - 1] + __expf(params[v * 3 + i]);
            const float C10[11] = {1.f,10.f,45.f,120.f,210.f,252.f,210.f,120.f,45.f,10.f,1.f};
            const float C9[10]  = {1.f,9.f,36.f,84.f,126.f,126.f,84.f,36.f,9.f,1.f};
#pragma unroll
            for (int q = 0; q < 2; ++q) {
                const int K = K0 + q;
                float val = 0.0f;
                if (ch < 3) {
                    if (K <= 10) val = C10[K] * pr[K];
                } else {
                    const int v = K - 16;
                    if (v >= 0 && v <= 9) val = 10.0f * C9[v] * (pr[v + 1] - pr[v]);
                }
                if (q == 0) v0 = val; else v1 = val;
            }
        }
        sA[row * 32 + K0]     = (_Float16)v0;
        sA[row * 32 + K0 + 1] = (_Float16)v1;
    }
    __syncthreads();

    const int lane = threadIdx.x & 31;
    const int m    = lane & 15;     // element column / A row
    const int h    = lane >> 4;     // wave half
    const bool hiHalf = (h != 0);

    // per-lane A fragment: half h of lane m holds K 16h..16h+15 contiguously
    v16h afrag;
    {
        const v8h lo = *(const v8h*)&sA[m * 32 + h * 16];
        const v8h hi = *(const v8h*)&sA[m * 32 + h * 16 + 8];
#pragma unroll
        for (int i = 0; i < 8; ++i) { afrag[i] = lo[i]; afrag[8 + i] = hi[i]; }
    }

    // normalization constants (polynomial_range: row0=lo, row1=hi)
    float r0[3], rs[3];
#pragma unroll
    for (int i = 0; i < 3; ++i) {
        r0[i] = prange[i];
        rs[i] = 1.0f / (prange[3 + i] - prange[i]);
    }
    const float ldglob = ldg_ptr[0];

    const int nwaves = (gridDim.x * blockDim.x) >> 5;
    const int wid    = (blockIdx.x * blockDim.x + threadIdx.x) >> 5;

    // ---- main loop: 2 groups (32 elements) per iteration, no guards ----
    const int kstep = (nwaves * 32) % 3;
    int k0 = (wid * 32 + m) % 3;
    for (int p = wid; p < gpair; p += nwaves) {
        const int e0 = p * 32 + m;
        const int e1 = e0 + 16;
        int k1i = k0 + 1; if (k1i >= 3) k1i -= 3;

        __builtin_prefetch(inp + (long)(p + nwaves) * 32 + m, 0, 0);

        const float xin0 = inp[e0];
        const float xin1 = inp[e1];

        const float r00 = (k0 == 2) ? r0[2] : ((k0 == 1) ? r0[1] : r0[0]);
        const float rs0 = (k0 == 2) ? rs[2] : ((k0 == 1) ? rs[1] : rs[0]);
        const float r01 = (k1i == 2) ? r0[2] : ((k1i == 1) ? r0[1] : r0[0]);
        const float rs1 = (k1i == 2) ? rs[2] : ((k1i == 1) ? rs[1] : rs[0]);

        // two independent WMMA chains -> scheduler fills D-read hazard slots
        const v8f d0 = bern_group((xin0 - r00) * rs0, hiHalf, afrag);
        const v8f d1 = bern_group((xin1 - r01) * rs1, hiHalf, afrag);

        const float val0 = bern_extract(d0, k0,  hiHalf, ldglob);
        const float val1 = bern_extract(d1, k1i, hiHalf, ldglob);

        const int idx0 = e0 + (hiHalf ? ntot : 0);   // [output | log_d] concat layout
        const int idx1 = e1 + (hiHalf ? ntot : 0);
        out[idx0] = val0;
        out[idx1] = val1;

        k0 += kstep; if (k0 >= 3) k0 -= 3;
    }

    // ---- guarded tail: remaining groups (only taken when ntot % 32 != 0) ----
    for (int g = gpair * 2 + wid; g < ngroups; g += nwaves) {
        const int e = g * 16 + m;
        const int k = e % 3;
        float xin = 0.5f;
        if (e < ntot) xin = inp[e];
        const float r0s = (k == 2) ? r0[2] : ((k == 1) ? r0[1] : r0[0]);
        const float rss = (k == 2) ? rs[2] : ((k == 1) ? rs[1] : rs[0]);
        const v8f d = bern_group((xin - r0s) * rss, hiHalf, afrag);
        const float val = bern_extract(d, k, hiHalf, ldglob);
        const int idx = e + (hiHalf ? ntot : 0);
        if (e < ntot) out[idx] = val;
    }
}

extern "C" void kernel_launch(void* const* d_in, const int* in_sizes, int n_in,
                              void* d_out, int out_size, void* d_ws, size_t ws_size,
                              hipStream_t stream) {
    const float* inp    = (const float*)d_in[0];   // [N, 3] f32
    const float* params = (const float*)d_in[1];   // [11, 3] f32
    const float* prange = (const float*)d_in[2];   // [2, 3] f32
    const float* ldg    = (const float*)d_in[3];   // scalar f32
    float* out          = (float*)d_out;           // [2 * N * 3] f32

    const int ntot    = in_sizes[0];               // N * 3 flat elements
    const int ngroups = (ntot + 15) / 16;          // 16-element groups (incl. partial)
    const int gpair   = (ntot / 16) / 2;           // pairs of FULL groups (unguarded path)

    int blocks = 2048;                             // 8 waves/block, persistent loop
    const int max_blocks = (ngroups + 7) / 8;
    if (blocks > max_blocks) blocks = max_blocks;
    if (blocks < 1) blocks = 1;

    Transformation_16509854286115_kernel<<<dim3(blocks), dim3(256), 0, stream>>>(
        inp, params, prange, ldg, out, ntot, ngroups, gpair);
}
